// Model_10926396801550
// MI455X (gfx1250) — compile-verified, matrix operand order checked by reading, not run on hardware
//
#include <hip/hip_runtime.h>
#include <math.h>

// ---------------------------------------------------------------------------
// CDNA5 (gfx1250) implementation. wave32. All dense contractions go through a
// generic strided/batched bf16 WMMA GEMM (v_wmma_f32_16x16x32_bf16, fp32 acc).
// Retrieval streams the 1.8GB database once (non-temporal b128 loads), fusing
// dot-products (WMMA) with db-row norm accumulation; LDS reduction + packed
// global argmax.  Hot loops are branch-free: unguarded vectorized main body +
// clamped-load tail (no EXEC-masked per-element loads).
// ---------------------------------------------------------------------------

typedef __attribute__((ext_vector_type(16))) __bf16 v16bf;
typedef __attribute__((ext_vector_type(8)))  float  v8f;
typedef __attribute__((ext_vector_type(4), aligned(4))) float float4u; // b128, align 4 ok
typedef long long i64;

#define D_MODEL 512
#define NHEAD   8
#define DKH     64
#define EFF     2048
#define NLAYERS 4
#define VOCAB   556
#define BB      16
#define CHN     6
#define LFR     1500
#define FIN     5
#define TTOK    64
#define NDBROW  10000
#define DDBCOL  45000
#define LIMU    45

// -------------------------------- device helpers ---------------------------
// WMMA 16-bit operand layout per lane (lane = half*16 + idx, row = row0+idx):
// v16 elements 0..7  = k = kb .. kb+7      (kb = k0 + half*8)
// v16 elements 8..15 = k = kb+16 .. kb+23

// Vectorized tile load, kStride==1, caller guarantees kb+23 < K(row length).
__device__ __forceinline__ v16bf tile_vec(const float* __restrict__ p, int kb, bool nt,
                                          float* nrm)
{
  const float4u* q = (const float4u*)(p + kb);
  float4u v0, v1, v2, v3;
  if (nt) {
    v0 = __builtin_nontemporal_load(q + 0);
    v1 = __builtin_nontemporal_load(q + 1);
    v2 = __builtin_nontemporal_load(q + 4);
    v3 = __builtin_nontemporal_load(q + 5);
  } else {
    v0 = q[0]; v1 = q[1]; v2 = q[4]; v3 = q[5];
  }
  v16bf r;
#pragma unroll
  for (int i = 0; i < 4; ++i) {
    r[i]      = (__bf16)v0[i];
    r[4 + i]  = (__bf16)v1[i];
    r[8 + i]  = (__bf16)v2[i];
    r[12 + i] = (__bf16)v3[i];
    if (nrm) *nrm += v0[i]*v0[i] + v1[i]*v1[i] + v2[i]*v2[i] + v3[i]*v3[i];
  }
  return r;
}

// Branch-free scalar tile load with K guard: unconditional clamped loads,
// value select to zero (v_cndmask, no EXEC divergence).
__device__ __forceinline__ v16bf tile_scalar(const float* __restrict__ p, i64 kStride,
                                             int K, int kb, float* nrm)
{
  v16bf r;
#pragma unroll
  for (int i = 0; i < 8; ++i) {
    int k  = kb + i;
    int k2 = kb + 16 + i;
    int kc  = (k  < K) ? k  : (K - 1);
    int kc2 = (k2 < K) ? k2 : (K - 1);
    float f  = p[(i64)kc  * kStride];
    float f2 = p[(i64)kc2 * kStride];
    f  = (k  < K) ? f  : 0.0f;
    f2 = (k2 < K) ? f2 : 0.0f;
    r[i]     = (__bf16)f;
    r[8 + i] = (__bf16)f2;
    if (nrm) *nrm += f * f + f2 * f2;
  }
  return r;
}

// ------------------------- generic batched WMMA GEMM -----------------------
// C[z][m][n] = act( scale * sum_k A[z][m][k] * B[z][n][k] + bias )
//   A element: A + zb0*sAb0 + zb1*sAb1 + m*sAm + k          (k contiguous)
//   B element: B + zb0*sBb0 + zb1*sBb1 + n*sBn + k*sBk      (i.e. B^T[n,k])
//   C element: C + zb0*sCb0 + zb1*sCb1 + m*sCm + n          (n contiguous)
//   z = blockIdx.z ; zb0 = z / batchInner ; zb1 = z % batchInner
// biasMode: 0 none, 1 bias[n], 2 bias[m].  act: 0 none, 1 exact GELU.
__global__ void wmma_gemm_kernel(const float* __restrict__ A, const float* __restrict__ B,
                                 const float* __restrict__ bias, float* __restrict__ C,
                                 int M, int N, int K,
                                 i64 sAm, i64 sAb0, i64 sAb1,
                                 i64 sBn, i64 sBk, i64 sBb0, i64 sBb1,
                                 i64 sCm, i64 sCb0, i64 sCb1,
                                 int batchInner, int biasMode, int act, float scale)
{
  const int z    = blockIdx.z;
  const int zb0  = z / batchInner;
  const int zb1  = z % batchInner;
  const int wave = threadIdx.x >> 5;
  const int lane = threadIdx.x & 31;
  const int half = lane >> 4;
  const int idx  = lane & 15;
  const int m0 = blockIdx.x * 16;
  const int n0 = (blockIdx.y * 4 + wave) * 16;
  if (m0 >= M || n0 >= N) return;                 // wave-uniform: EXEC stays full

  const float* Ab = A + (i64)zb0 * sAb0 + (i64)zb1 * sAb1;
  const float* Bb = B + (i64)zb0 * sBb0 + (i64)zb1 * sBb1;
  float*       Cb = C + (i64)zb0 * sCb0 + (i64)zb1 * sCb1;

  int ra = m0 + idx; if (ra >= M) ra = M - 1;     // clamp: OOB rows masked at store
  int rb = n0 + idx; if (rb >= N) rb = N - 1;
  const float* pA = Ab + (i64)ra * sAm;
  const float* pB = Bb + (i64)rb * sBn;

  const int Kmain = K & ~31;
  v8f acc = {};
  if (sBk == 1) {                                 // uniform branch
    for (int k0 = 0; k0 < Kmain; k0 += 32) {
      v16bf a = tile_vec(pA, k0 + half * 8, false, nullptr);
      v16bf b = tile_vec(pB, k0 + half * 8, false, nullptr);
      acc = __builtin_amdgcn_wmma_f32_16x16x32_bf16(false, a, false, b, (short)0, acc,
                                                    false, false);
    }
  } else {
    for (int k0 = 0; k0 < Kmain; k0 += 32) {
      v16bf a = tile_vec(pA, k0 + half * 8, false, nullptr);
      v16bf b = tile_scalar(pB, sBk, K, k0 + half * 8, nullptr);
      acc = __builtin_amdgcn_wmma_f32_16x16x32_bf16(false, a, false, b, (short)0, acc,
                                                    false, false);
    }
  }
  if (Kmain < K) {                                // single guarded tail step
    v16bf a = tile_scalar(pA, 1,   K, Kmain + half * 8, nullptr);
    v16bf b = tile_scalar(pB, sBk, K, Kmain + half * 8, nullptr);
    acc = __builtin_amdgcn_wmma_f32_16x16x32_bf16(false, a, false, b, (short)0, acc,
                                                  false, false);
  }

  const int n = n0 + idx;
  if (n < N) {
#pragma unroll
    for (int r = 0; r < 8; ++r) {
      int m = m0 + r + half * 8;
      if (m < M) {
        float v = acc[r] * scale;
        if (biasMode == 1) v += bias[n];
        else if (biasMode == 2) v += bias[m];
        if (act == 1) v = 0.5f * v * (1.0f + erff(v * 0.70710678118654752f));
        Cb[(i64)m * sCm + n] = v;
      }
    }
  }
}

// ------------------------------ retrieval ----------------------------------
// One block per 16 db rows; 8 waves split K=45000.  Each wave accumulates a
// 16x16 dot tile with WMMA (bf16 in, fp32 acc) AND the db-row sum of squares,
// LDS-reduces across waves, then does a global packed-key argmax.
// db rows are 16B-aligned (45000*4 % 16 == 0): fully vectorized NT b128 loads.
__global__ void retrieval_kernel(const float* __restrict__ imu,  // [16,45000]
                                 const float* __restrict__ db,   // [10000,45000]
                                 unsigned long long* __restrict__ keys) // [16]
{
  __shared__ float sC[16][16];
  __shared__ float sN[16];
  const int tid = threadIdx.x;                    // 256 threads = 8 waves
  if (tid < 16) sN[tid] = 0.0f;
  ((float*)sC)[tid] = 0.0f;
  __syncthreads();

  const int wave = tid >> 5, lane = tid & 31;
  const int half = lane >> 4, idx = lane & 15;
  const int n0 = blockIdx.x * 16;
  const float* arow = imu + (i64)idx * DDBCOL;
  const float* brow = db  + (i64)(n0 + idx) * DDBCOL;

  const int kchunk = 5632;                        // ceil(45000/8/32)*32
  const int ks = wave * kchunk;
  const int ke = (ks + kchunk < DDBCOL) ? (ks + kchunk) : DDBCOL;
  const int keMain = ks + ((ke - ks) & ~31);

  v8f acc = {};
  float nrm = 0.0f;
  for (int k0 = ks; k0 < keMain; k0 += 32) {
    __builtin_prefetch(brow + k0 + 2048, 0, 3);
    v16bf a = tile_vec(arow, k0 + half * 8, false, nullptr);
    v16bf b = tile_vec(brow, k0 + half * 8, true,  &nrm);
    acc = __builtin_amdgcn_wmma_f32_16x16x32_bf16(false, a, false, b, (short)0, acc,
                                                  false, false);
  }
  if (keMain < ke) {                              // guarded tail (wave 7 only)
    v16bf a = tile_scalar(arow, 1, DDBCOL, keMain + half * 8, nullptr);
    v16bf b = tile_scalar(brow, 1, DDBCOL, keMain + half * 8, &nrm);
    acc = __builtin_amdgcn_wmma_f32_16x16x32_bf16(false, a, false, b, (short)0, acc,
                                                  false, false);
  }
#pragma unroll
  for (int r = 0; r < 8; ++r) atomicAdd(&sC[r + half * 8][idx], acc[r]);
  atomicAdd(&sN[idx], nrm);
  __syncthreads();

  if (tid < 16) {
    const int m = tid;                            // batch row
    float best = -3.4e38f; int bj = 0;
    for (int nn = 0; nn < 16; ++nn) {
      float c = sC[m][nn] / fmaxf(sqrtf(sN[nn]), 1e-8f);
      if (c > best) { best = c; bj = n0 + nn; }
    }
    unsigned int fb  = __float_as_uint(best);
    unsigned int ord = (fb & 0x80000000u) ? ~fb : (fb | 0x80000000u);
    unsigned long long key =
        ((unsigned long long)ord << 32) | (unsigned long long)(0xFFFFFFFFu - (unsigned)bj);
    atomicMax(&keys[m], key);                     // first-index wins on ties
  }
}

__global__ void init_keys_kernel(unsigned long long* keys)
{
  if (threadIdx.x < BB) keys[threadIdx.x] = 0ull;
}

// ------------------------------- softmax -----------------------------------
__global__ void softmax_kernel(float* __restrict__ S, int Lq, int Lk, int causal)
{
  const i64 row = blockIdx.x;
  float* s = S + row * (i64)Lk;
  const int i   = (int)(row % Lq);
  const int lim = causal ? (i + 1) : Lk;
  const int lane = threadIdx.x;
  float m = -3.4e38f;
  for (int j = lane; j < lim; j += 32) m = fmaxf(m, s[j]);
  for (int off = 16; off > 0; off >>= 1) m = fmaxf(m, __shfl_xor(m, off, 32));
  float sum = 0.0f;
  for (int j = lane; j < lim; j += 32) { float e = expf(s[j] - m); s[j] = e; sum += e; }
  for (int off = 16; off > 0; off >>= 1) sum += __shfl_xor(sum, off, 32);
  float inv = 1.0f / sum;
  for (int j = lane; j < lim; j += 32) s[j] *= inv;
  for (int j = lim + lane; j < Lk; j += 32) s[j] = 0.0f;   // masked -> prob 0
}

// ------------------------- residual-add + layernorm ------------------------
__global__ void add_ln_kernel(const float* __restrict__ a, const float* __restrict__ b,
                              const float* __restrict__ g, const float* __restrict__ beta,
                              float* __restrict__ out)
{
  const int row = blockIdx.x;
  const int tid = threadIdx.x;                   // 256, D=512 -> 2 per thread
  const i64 base = (i64)row * D_MODEL;
  float x0 = a[base + tid] + b[base + tid];
  float x1 = a[base + tid + 256] + b[base + tid + 256];
  float s = x0 + x1, q = x0 * x0 + x1 * x1;
  for (int off = 16; off > 0; off >>= 1) {
    s += __shfl_xor(s, off, 32);
    q += __shfl_xor(q, off, 32);
  }
  __shared__ float red[18];
  const int wv = tid >> 5, ln = tid & 31;
  if (ln == 0) { red[wv] = s; red[wv + 8] = q; }
  __syncthreads();
  if (tid == 0) {
    float ts = 0.0f, tq = 0.0f;
    for (int w = 0; w < 8; ++w) { ts += red[w]; tq += red[w + 8]; }
    red[16] = ts; red[17] = tq;
  }
  __syncthreads();
  float mean = red[16] * (1.0f / D_MODEL);
  float var  = red[17] * (1.0f / D_MODEL) - mean * mean;
  float inv  = rsqrtf(var + 1e-5f);
  out[base + tid]       = (x0 - mean) * inv * g[tid]       + beta[tid];
  out[base + tid + 256] = (x1 - mean) * inv * g[tid + 256] + beta[tid + 256];
}

// ----------------------------- small data movers ---------------------------
__global__ void transpose_src_kernel(const float* __restrict__ src, float* __restrict__ out)
{
  i64 i = (i64)blockIdx.x * blockDim.x + threadIdx.x;   // 16*1500*30
  if (i >= (i64)BB * LFR * (CHN * FIN)) return;
  int c = (int)(i % (CHN * FIN)); i64 t = i / (CHN * FIN);
  int l = (int)(t % LFR); int b = (int)(t / LFR);
  int ch = c / FIN, f = c % FIN;
  out[i] = src[(((i64)b * CHN + ch) * LFR + l) * FIN + f];
}

__global__ void gather_addinfo_kernel(const unsigned long long* __restrict__ keys,
                                      const int* __restrict__ dby,
                                      const float* __restrict__ emb,
                                      float* __restrict__ fus)   // [16,90,512], rows 45..89
{
  const int b = blockIdx.x, t = blockIdx.y, tid = threadIdx.x;
  const unsigned idxb = 0xFFFFFFFFu - (unsigned)(keys[b] & 0xFFFFFFFFull);
  const int token = dby[(i64)idxb * LIMU + t];
  const float* s = emb + (i64)token * D_MODEL;
  float* d = fus + ((i64)b * (2 * LIMU) + LIMU + t) * D_MODEL;
  d[tid] = s[tid]; d[tid + 256] = s[tid + 256];
}

__global__ void gather_tgt_kernel(const int* __restrict__ tgt, const float* __restrict__ emb,
                                  float* __restrict__ wemb)      // [16,64,512]
{
  const int b = blockIdx.x, t = blockIdx.y, tid = threadIdx.x;
  const int token = tgt[b * TTOK + t];
  const float* s = emb + (i64)token * D_MODEL;
  float* d = wemb + ((i64)b * TTOK + t) * D_MODEL;
  d[tid] = s[tid]; d[tid + 256] = s[tid + 256];
}

__global__ void copy_to_fus_kernel(const float* __restrict__ in, float* __restrict__ fus)
{
  const int b = blockIdx.x, t = blockIdx.y, tid = threadIdx.x;  // 16 x 45
  const float* s = in + ((i64)b * LIMU + t) * D_MODEL;
  float* d = fus + ((i64)b * (2 * LIMU) + t) * D_MODEL;
  d[tid] = s[tid]; d[tid + 256] = s[tid + 256];
}

// ------------------------------- host side ---------------------------------

struct Attn {
  const float *wq_w, *wq_b, *wk_w, *wk_b, *wv_w, *wv_b, *po_w, *po_b;
};
struct EncLayer {
  Attn at;
  const float *ln1_g, *ln1_b, *ln2_g, *ln2_b, *f1_w, *f1_b, *f2_w, *f2_b;
};

static void launch_gemm(hipStream_t st, const float* A, const float* B, const float* bias,
                        float* C, int M, int N, int K,
                        i64 sAm, i64 sAb0, i64 sAb1,
                        i64 sBn, i64 sBk, i64 sBb0, i64 sBb1,
                        i64 sCm, i64 sCb0, i64 sCb1,
                        int nbatch, int batchInner, int biasMode, int act, float scale)
{
  dim3 grid((M + 15) / 16, (N + 63) / 64, nbatch);
  hipLaunchKernelGGL(wmma_gemm_kernel, grid, dim3(128), 0, st, A, B, bias, C, M, N, K,
                     sAm, sAb0, sAb1, sBn, sBk, sBb0, sBb1, sCm, sCb0, sCb1,
                     batchInner, biasMode, act, scale);
}

// q:[16,Lq,512]  ksrc/vsrc:[16,Lk,512]
static void run_mha(hipStream_t st, const float* q, int Lq, const float* ksrc,
                    const float* vsrc, int Lk, const Attn& p, int causal, float* out,
                    float* qh, float* kh, float* vh, float* scores)
{
  const int Mq = BB * Lq, Mk = BB * Lk;
  const i64 D = D_MODEL;
  launch_gemm(st, q,    p.wq_w, p.wq_b, qh, Mq, D_MODEL, D_MODEL,
              D,0,0,  D,1,0,0,  D,0,0,  1,1, 1,0, 1.0f);
  launch_gemm(st, ksrc, p.wk_w, p.wk_b, kh, Mk, D_MODEL, D_MODEL,
              D,0,0,  D,1,0,0,  D,0,0,  1,1, 1,0, 1.0f);
  launch_gemm(st, vsrc, p.wv_w, p.wv_b, vh, Mk, D_MODEL, D_MODEL,
              D,0,0,  D,1,0,0,  D,0,0,  1,1, 1,0, 1.0f);
  // scores[b,h,i,j] = qh . kh / sqrt(64)
  launch_gemm(st, qh, kh, nullptr, scores, Lq, Lk, DKH,
              D, (i64)Lq * D, DKH,
              D, 1, (i64)Lk * D, DKH,
              Lk, (i64)NHEAD * Lq * Lk, (i64)Lq * Lk,
              BB * NHEAD, NHEAD, 0, 0, 0.125f);
  hipLaunchKernelGGL(softmax_kernel, dim3(BB * NHEAD * Lq), dim3(32), 0, st,
                     scores, Lq, Lk, causal);
  // attn[b,i,h,d] = P @ V   (reuse kh as output)
  launch_gemm(st, scores, vh, nullptr, kh, Lq, DKH, Lk,
              Lk, (i64)NHEAD * Lq * Lk, (i64)Lq * Lk,
              1, D, (i64)Lk * D, DKH,
              D, (i64)Lq * D, DKH,
              BB * NHEAD, NHEAD, 0, 0, 1.0f);
  launch_gemm(st, kh, p.po_w, p.po_b, out, Mq, D_MODEL, D_MODEL,
              D,0,0,  D,1,0,0,  D,0,0,  1,1, 1,0, 1.0f);
}

static void run_block(hipStream_t st, const float* qk, int L, const float* vin,
                      const EncLayer& lp, int causal, float* vout,
                      float* qh, float* kh, float* vh, float* scores,
                      float* mhaout, float* xln, float* ffn)
{
  const int rows = BB * L;
  run_mha(st, qk, L, qk, vin, L, lp.at, causal, mhaout, qh, kh, vh, scores);
  hipLaunchKernelGGL(add_ln_kernel, dim3(rows), dim3(256), 0, st,
                     mhaout, qk, lp.ln1_g, lp.ln1_b, xln);
  launch_gemm(st, xln, lp.f1_w, lp.f1_b, ffn, rows, EFF, D_MODEL,
              D_MODEL,0,0, D_MODEL,1,0,0, EFF,0,0, 1,1, 1,1, 1.0f);      // GELU
  launch_gemm(st, ffn, lp.f2_w, lp.f2_b, mhaout, rows, D_MODEL, EFF,
              EFF,0,0, EFF,1,0,0, D_MODEL,0,0, 1,1, 1,0, 1.0f);
  hipLaunchKernelGGL(add_ln_kernel, dim3(rows), dim3(256), 0, st,
                     xln, mhaout, lp.ln2_g, lp.ln2_b, vout);
}

// -------- input index map (top-level insertion order; params leaves sorted) -
enum {
  IN_SRC = 0, IN_TGT = 1, IN_DBX = 2, IN_DBY = 3,
  IN_EMB = 4,
  IN_FUS_PO_B = 5, IN_FUS_PO_W, IN_FUS_WK_B, IN_FUS_WK_W,
  IN_FUS_WQ_B, IN_FUS_WQ_W, IN_FUS_WV_B, IN_FUS_WV_W,
  IN_IMU_BASE = 13,   // f1_b,f1_w,f2_b,f2_w,ln1_b,ln1_g,ln2_b,ln2_g,po_b,po_w,wk_b,wk_w,wq_b,wq_w,wv_b,wv_w
  IN_L1_B = 29, IN_L1_W = 30, IN_L2_B = 31, IN_L2_W = 32,
  IN_L3_B = 33, IN_L3_W = 34, IN_PROJ_B = 35, IN_PROJ_W = 36,
  IN_TXT_BASE = 37
};

static EncLayer enc_layer(void* const* din, int base, int i)
{
  const float* P[16];
  for (int j = 0; j < 16; ++j) P[j] = (const float*)din[base + j];
  EncLayer L;
  L.f1_b     = P[0]  + (i64)i * EFF;
  L.f1_w     = P[1]  + (i64)i * EFF * D_MODEL;
  L.f2_b     = P[2]  + (i64)i * D_MODEL;
  L.f2_w     = P[3]  + (i64)i * D_MODEL * EFF;
  L.ln1_b    = P[4]  + (i64)i * D_MODEL;
  L.ln1_g    = P[5]  + (i64)i * D_MODEL;
  L.ln2_b    = P[6]  + (i64)i * D_MODEL;
  L.ln2_g    = P[7]  + (i64)i * D_MODEL;
  L.at.po_b  = P[8]  + (i64)i * D_MODEL;
  L.at.po_w  = P[9]  + (i64)i * D_MODEL * D_MODEL;
  L.at.wk_b  = P[10] + (i64)i * D_MODEL;
  L.at.wk_w  = P[11] + (i64)i * D_MODEL * D_MODEL;
  L.at.wq_b  = P[12] + (i64)i * D_MODEL;
  L.at.wq_w  = P[13] + (i64)i * D_MODEL * D_MODEL;
  L.at.wv_b  = P[14] + (i64)i * D_MODEL;
  L.at.wv_w  = P[15] + (i64)i * D_MODEL * D_MODEL;
  return L;
}

extern "C" void kernel_launch(void* const* d_in, const int* in_sizes, int n_in,
                              void* d_out, int out_size, void* d_ws, size_t ws_size,
                              hipStream_t stream)
{
  (void)in_sizes; (void)n_in; (void)out_size; (void)ws_size;
  const float* src  = (const float*)d_in[IN_SRC];
  const int*   tgt  = (const int*)  d_in[IN_TGT];
  const float* dbx  = (const float*)d_in[IN_DBX];
  const int*   dby  = (const int*)  d_in[IN_DBY];
  const float* emb  = (const float*)d_in[IN_EMB];

  // ----- workspace bump allocator (floats; keys first for u64 alignment) ---
  unsigned long long* keys = (unsigned long long*)d_ws;
  float* base = (float*)d_ws;
  size_t off = 32;                                   // skip 128B for keys
  auto alloc = [&](size_t n) { float* r = base + off; off += (n + 63) & ~(size_t)63; return r; };

  float* srcT   = alloc((size_t)BB * LFR * (CHN * FIN));   // 720000
  float* x1     = alloc((size_t)BB * LIMU * (CHN * FIN));  // 21600
  float* x2     = alloc((size_t)BB * LIMU * D_MODEL);      // 368640
  float* vA     = alloc((size_t)BB * LIMU * D_MODEL);
  float* vB     = alloc((size_t)BB * LIMU * D_MODEL);
  float* wemb   = alloc((size_t)BB * TTOK * D_MODEL);      // 524288
  float* tA     = alloc((size_t)BB * TTOK * D_MODEL);
  float* tB     = alloc((size_t)BB * TTOK * D_MODEL);
  float* fus    = alloc((size_t)BB * 2 * LIMU * D_MODEL);  // 737280
  float* fus2   = alloc((size_t)BB * LIMU * D_MODEL);
  float* qh     = alloc((size_t)BB * TTOK * D_MODEL);
  float* kh     = alloc((size_t)BB * TTOK * D_MODEL);
  float* vh     = alloc((size_t)BB * TTOK * D_MODEL);
  float* scores = alloc((size_t)BB * NHEAD * TTOK * TTOK); // 524288
  float* mhaout = alloc((size_t)BB * TTOK * D_MODEL);
  float* xln    = alloc((size_t)BB * TTOK * D_MODEL);
  float* ffn    = alloc((size_t)BB * TTOK * EFF);          // 2097152

  // ----- retrieval: single 1.8GB streaming pass + global argmax ------------
  hipLaunchKernelGGL(init_keys_kernel, dim3(1), dim3(32), 0, stream, keys);
  hipLaunchKernelGGL(retrieval_kernel, dim3(NDBROW / 16), dim3(256), 0, stream,
                     src, dbx, keys);

  // ----- IMU branch --------------------------------------------------------
  hipLaunchKernelGGL(transpose_src_kernel,
                     dim3(((size_t)BB * LFR * CHN * FIN + 255) / 256), dim3(256), 0, stream,
                     src, srcT);
  // x1[b,o,c] = sum_l l1_w[o,l] * srcT[b,l,c] + l1_b[o]
  launch_gemm(stream, (const float*)d_in[IN_L1_W], srcT, (const float*)d_in[IN_L1_B], x1,
              LIMU, CHN * FIN, LFR,
              LFR, 0, 0,
              1, CHN * FIN, (i64)LFR * CHN * FIN, 0,
              CHN * FIN, (i64)LIMU * CHN * FIN, 0,
              BB, 1, 2, 0, 1.0f);
  // x2 = x1 @ l2_w^T + l2_b   (K=30, zero-padded to 32)
  launch_gemm(stream, x1, (const float*)d_in[IN_L2_W], (const float*)d_in[IN_L2_B], x2,
              BB * LIMU, D_MODEL, CHN * FIN,
              CHN * FIN, 0, 0,
              CHN * FIN, 1, 0, 0,
              D_MODEL, 0, 0,
              1, 1, 1, 0, 1.0f);
  {
    const float* vin = x2;
    float* bufs[2] = { vA, vB };
    for (int i = 0; i < NLAYERS; ++i) {
      EncLayer L = enc_layer(d_in, IN_IMU_BASE, i);
      float* vout = bufs[i & 1];
      run_block(stream, x2, LIMU, vin, L, 0, vout, qh, kh, vh, scores, mhaout, xln, ffn);
      vin = vout;
    }
  }
  hipLaunchKernelGGL(copy_to_fus_kernel, dim3(BB, LIMU), dim3(256), 0, stream, vB, fus);
  hipLaunchKernelGGL(gather_addinfo_kernel, dim3(BB, LIMU), dim3(256), 0, stream,
                     keys, dby, emb, fus);
  // fus2[b,o,d] = sum_c l3_w[o,c] * fus[b,c,d] + l3_b[o]
  launch_gemm(stream, (const float*)d_in[IN_L3_W], fus, (const float*)d_in[IN_L3_B], fus2,
              LIMU, D_MODEL, 2 * LIMU,
              2 * LIMU, 0, 0,
              1, D_MODEL, (i64)2 * LIMU * D_MODEL, 0,
              D_MODEL, (i64)LIMU * D_MODEL, 0,
              BB, 1, 2, 0, 1.0f);

  // ----- text branch (causal) ---------------------------------------------
  hipLaunchKernelGGL(gather_tgt_kernel, dim3(BB, TTOK), dim3(256), 0, stream,
                     tgt, emb, wemb);
  {
    const float* vin = wemb;
    float* bufs[2] = { tA, tB };
    for (int i = 0; i < NLAYERS; ++i) {
      EncLayer L = enc_layer(d_in, IN_TXT_BASE, i);
      float* vout = bufs[i & 1];
      run_block(stream, wemb, TTOK, vin, L, 1, vout, qh, kh, vh, scores, mhaout, xln, ffn);
      vin = vout;
    }
  }

  // ----- fusion cross-attention + vocab projection -------------------------
  Attn fp;
  fp.wq_w = (const float*)d_in[IN_FUS_WQ_W]; fp.wq_b = (const float*)d_in[IN_FUS_WQ_B];
  fp.wk_w = (const float*)d_in[IN_FUS_WK_W]; fp.wk_b = (const float*)d_in[IN_FUS_WK_B];
  fp.wv_w = (const float*)d_in[IN_FUS_WV_W]; fp.wv_b = (const float*)d_in[IN_FUS_WV_B];
  fp.po_w = (const float*)d_in[IN_FUS_PO_W]; fp.po_b = (const float*)d_in[IN_FUS_PO_B];
  run_mha(stream, tB, TTOK, fus2, fus2, LIMU, fp, 0, mhaout, qh, kh, vh, scores);
  launch_gemm(stream, mhaout, (const float*)d_in[IN_PROJ_W], (const float*)d_in[IN_PROJ_B],
              (float*)d_out, BB * TTOK, VOCAB, D_MODEL,
              D_MODEL, 0, 0,
              D_MODEL, 1, 0, 0,
              VOCAB, 0, 0,
              1, 1, 1, 0, 1.0f);
}
